// MaskUnitAttention_40467181862987
// MI455X (gfx1250) — matrix operand order repaired and checked by reading, not compiled, not run on hardware
//
#include <hip/hip_runtime.h>
#include <hip/hip_bf16.h>
#include <math.h>

typedef __bf16 bf16;
typedef __bf16 v16bf __attribute__((ext_vector_type(16)));
typedef __bf16 v8bf  __attribute__((ext_vector_type(8)));
typedef float  v8f   __attribute__((ext_vector_type(8)));

#define DIM      768
#define HEADS    12
#define HEAD_DIM 64
#define NTOK     16384
#define BATCH    4
#define NW       64      // num windows
#define TPW      256     // tokens per window
#define QKV_N    2304
#define M_QKV    (BATCH*NTOK)   // 65536
#define M_PROJ   (BATCH*4096)   // 16384

__device__ __forceinline__ v16bf join8(v8bf lo, v8bf hi) {
  return __builtin_shufflevector(lo, hi, 0,1,2,3,4,5,6,7,8,9,10,11,12,13,14,15);
}
__device__ __forceinline__ v8f zero8() {
  v8f z;
#pragma unroll
  for (int i = 0; i < 8; ++i) z[i] = 0.0f;
  return z;
}
__device__ __forceinline__ v8f wmma_bf16(v16bf a, v16bf b, v8f c) {
  return __builtin_amdgcn_wmma_f32_16x16x32_bf16(false, a, false, b, (short)0, c, false, false);
}

// ---------------------------------------------------------------- convert f32 -> bf16
__global__ __launch_bounds__(256) void cvt_f32_bf16(const float* __restrict__ src,
                                                    bf16* __restrict__ dst, int n4) {
  int i = blockIdx.x * blockDim.x + threadIdx.x;
  if (i < n4) {
    float4 f = reinterpret_cast<const float4*>(src)[i];
    union { bf16 b[4]; uint2 u; } p;
    p.b[0] = (bf16)f.x; p.b[1] = (bf16)f.y; p.b[2] = (bf16)f.z; p.b[3] = (bf16)f.w;
    reinterpret_cast<uint2*>(dst)[i] = p.u;
  }
}

// ---------------------------------------------------------------- C[M,N] = A[M,K] @ Bw[N,K]^T + bias
// 128x128x32 tiles, 256 threads = 8 waves, each wave: 16 rows x 128 cols.
// Double-buffered LDS: global loads for tile k+1 issued before the WMMA block
// of tile k, one barrier per K-step.
template <bool OUT_BF16>
__global__ __launch_bounds__(256) void gemm_bf16_tn(const bf16* __restrict__ A,
                                                    const bf16* __restrict__ Bw,
                                                    const float* __restrict__ bias,
                                                    void* __restrict__ Cout,
                                                    int M, int N, int K) {
  constexpr int STR = 40;                 // padded LDS row stride (bank-conflict free)
  __shared__ bf16 As[2][128 * STR];
  __shared__ bf16 Bs[2][128 * STR];

  const int tid  = threadIdx.x;
  const int lane = tid & 31;
  const int wave = tid >> 5;              // 0..7
  const int lrow = lane & 15;
  const int lhi  = lane >> 4;             // 0/1
  const int m0   = blockIdx.x * 128;
  const int n0   = blockIdx.y * 128;

  v8f acc[8];
#pragma unroll
  for (int t = 0; t < 8; ++t) acc[t] = zero8();

  const int ldrow = tid >> 1;             // 0..127
  const int ldk   = (tid & 1) * 16;       // 0/16
  const bf16* ga = A  + (size_t)(m0 + ldrow) * K + ldk;
  const bf16* gb = Bw + (size_t)(n0 + ldrow) * K + ldk;
  const int lds_off = ldrow * STR + ldk;

  // prologue: fill buffer 0
  {
    uint4 a0 = *reinterpret_cast<const uint4*>(ga);
    uint4 a1 = *reinterpret_cast<const uint4*>(ga + 8);
    uint4 b0 = *reinterpret_cast<const uint4*>(gb);
    uint4 b1 = *reinterpret_cast<const uint4*>(gb + 8);
    *reinterpret_cast<uint4*>(&As[0][lds_off])     = a0;
    *reinterpret_cast<uint4*>(&As[0][lds_off + 8]) = a1;
    *reinterpret_cast<uint4*>(&Bs[0][lds_off])     = b0;
    *reinterpret_cast<uint4*>(&Bs[0][lds_off + 8]) = b1;
  }
  __syncthreads();

  const int nk = K / 32;
  for (int kt = 0; kt < nk; ++kt) {
    const int cur  = kt & 1;
    const bool more = (kt + 1) < nk;
    uint4 na0{}, na1{}, nb0{}, nb1{};
    if (more) {                            // issue next tile's global loads early
      const bf16* pa = ga + (kt + 1) * 32;
      const bf16* pb = gb + (kt + 1) * 32;
      na0 = *reinterpret_cast<const uint4*>(pa);
      na1 = *reinterpret_cast<const uint4*>(pa + 8);
      nb0 = *reinterpret_cast<const uint4*>(pb);
      nb1 = *reinterpret_cast<const uint4*>(pb + 8);
    }
    // A fragment (16x32): per-lane row = lane%16, K = lhi*16 .. +15
    const int arow = wave * 16 + lrow;
    v8bf alo = *reinterpret_cast<const v8bf*>(&As[cur][arow * STR + lhi * 16]);
    v8bf ahi = *reinterpret_cast<const v8bf*>(&As[cur][arow * STR + lhi * 16 + 8]);
    v16bf af = join8(alo, ahi);
#pragma unroll
    for (int t = 0; t < 8; ++t) {
      // B fragment (32x16, from Bw rows): per-lane row = tile*16 + lane%16, K = lhi*16 .. +15
      const int brow = t * 16 + lrow;
      v8bf blo = *reinterpret_cast<const v8bf*>(&Bs[cur][brow * STR + lhi * 16]);
      v8bf bhi = *reinterpret_cast<const v8bf*>(&Bs[cur][brow * STR + lhi * 16 + 8]);
      acc[t] = wmma_bf16(af, join8(blo, bhi), acc[t]);
    }
    if (more) {
      const int nxt = cur ^ 1;
      *reinterpret_cast<uint4*>(&As[nxt][lds_off])     = na0;
      *reinterpret_cast<uint4*>(&As[nxt][lds_off + 8]) = na1;
      *reinterpret_cast<uint4*>(&Bs[nxt][lds_off])     = nb0;
      *reinterpret_cast<uint4*>(&Bs[nxt][lds_off + 8]) = nb1;
      __syncthreads();
    }
  }

#pragma unroll
  for (int t = 0; t < 8; ++t) {
    const int n  = n0 + t * 16 + lrow;
    const float bv = bias[n];
#pragma unroll
    for (int r = 0; r < 8; ++r) {
      const int m = m0 + wave * 16 + r + lhi * 8;
      float val = acc[t][r] + bv;
      if constexpr (OUT_BF16)
        reinterpret_cast<bf16*>(Cout)[(size_t)m * N + n] = (bf16)val;
      else
        reinterpret_cast<float*>(Cout)[(size_t)m * N + n] = val;
    }
  }
}

// ---------------------------------------------------------------- windowed attention
// One block (8 waves, 256 threads) per (b, h, w).
// Token n = t*64 + w; feature = comp*768 + h*64 + d.
// Waves pair up on a 16-row q strip: waves 2s,2s+1 own strip s and split the
// score n-tiles (8 each) and the AV d-tiles (2 each).
__global__ __launch_bounds__(256) void mua_attn(const bf16* __restrict__ qkvb,
                                                bf16* __restrict__ ob) {
  extern __shared__ char smem[];
  bf16*  qp = (bf16*)smem;            // [64][72]  pooled+scaled q
  bf16*  kb = qp + 64 * 72;           // [256][72] k row-major
  bf16*  vT = kb + 256 * 72;          // [64][264] v transposed (d-major)
  float* sc = (float*)(vT + 64 * 264);// [64][264] scores f32
  bf16*  pr = (bf16*)(sc + 64 * 264); // [64][264] probs bf16

  const int blk = blockIdx.x;
  const int b = blk / (HEADS * NW);
  const int h = (blk / NW) % HEADS;
  const int w = blk % NW;

  const int tid  = threadIdx.x;
  const int lane = tid & 31;
  const int wave = tid >> 5;          // 0..7
  const int lrow = lane & 15;
  const int lhi  = lane >> 4;

  const size_t rstride = (size_t)NW * QKV_N; // stride between in-window tokens
  const bf16* base = qkvb + ((size_t)b * NTOK + w) * QKV_N + h * HEAD_DIM;

  // ---- phase 0a: load q, max-pool over 4 stride groups, fold 1/sqrt(hd)
  {
    const int qt = tid >> 2;            // 0..63
    const int d0 = (tid & 3) * 16;      // 0/16/32/48
    float mx[16];
#pragma unroll
    for (int e = 0; e < 16; ++e) mx[e] = -3.0e38f;
#pragma unroll
    for (int s = 0; s < 4; ++s) {
      const bf16* p = base + (size_t)(s * 64 + qt) * rstride + d0;
#pragma unroll
      for (int c = 0; c < 2; ++c) {
        v8bf vch = *reinterpret_cast<const v8bf*>(p + c * 8);
#pragma unroll
        for (int e = 0; e < 8; ++e)
          mx[c * 8 + e] = fmaxf(mx[c * 8 + e], (float)vch[e]);
      }
    }
#pragma unroll
    for (int e = 0; e < 16; ++e)
      qp[qt * 72 + d0 + e] = (bf16)(mx[e] * 0.125f);
  }

  // ---- phase 0b: stage k row-major, v transposed (one token row per thread)
  {
    const int kt = tid;                 // 0..255
    const bf16* pk = base + 768  + (size_t)kt * rstride;
    const bf16* pv = base + 1536 + (size_t)kt * rstride;
#pragma unroll
    for (int c = 0; c < 8; ++c) {
      v8bf kc = *reinterpret_cast<const v8bf*>(pk + c * 8);
      *reinterpret_cast<v8bf*>(&kb[kt * 72 + c * 8]) = kc;
      v8bf vc = *reinterpret_cast<const v8bf*>(pv + c * 8);
#pragma unroll
      for (int e = 0; e < 8; ++e)
        vT[(c * 8 + e) * 264 + kt] = vc[e];
    }
  }
  __syncthreads();

  // ---- phase 1: scores[64,256] = qp @ k^T  (K=64 -> 2 k-steps; 8 n-tiles per wave)
  {
    const int m0 = (wave >> 1) * 16;
    const int ntbase = (wave & 1) * 8;
    v16bf af[2];
#pragma unroll
    for (int ks = 0; ks < 2; ++ks) {
      const int ak = ks * 32 + lhi * 16;
      v8bf lo = *reinterpret_cast<const v8bf*>(&qp[(m0 + lrow) * 72 + ak]);
      v8bf hi = *reinterpret_cast<const v8bf*>(&qp[(m0 + lrow) * 72 + ak + 8]);
      af[ks] = join8(lo, hi);
    }
#pragma unroll
    for (int ti = 0; ti < 8; ++ti) {
      const int nt = ntbase + ti;
      v8f acc = zero8();
#pragma unroll
      for (int ks = 0; ks < 2; ++ks) {
        const int brow = nt * 16 + lrow;
        const int bk = ks * 32 + lhi * 16;
        v8bf lo = *reinterpret_cast<const v8bf*>(&kb[brow * 72 + bk]);
        v8bf hi = *reinterpret_cast<const v8bf*>(&kb[brow * 72 + bk + 8]);
        acc = wmma_bf16(af[ks], join8(lo, hi), acc);
      }
#pragma unroll
      for (int r = 0; r < 8; ++r)
        sc[(m0 + r + lhi * 8) * 264 + nt * 16 + lrow] = acc[r];
    }
  }
  __syncthreads();

  // ---- phase 2: softmax (4 threads per row, two-step shfl_xor reduce)
  {
    const int row = tid >> 2;
    float* srow = &sc[row * 264 + (tid & 3) * 64];
    float m = -3.0e38f;
#pragma unroll 4
    for (int j = 0; j < 64; ++j) m = fmaxf(m, srow[j]);
    m = fmaxf(m, __shfl_xor(m, 1, 32));
    m = fmaxf(m, __shfl_xor(m, 2, 32));
    float sum = 0.0f;
#pragma unroll 4
    for (int j = 0; j < 64; ++j) {
      float ev = __expf(srow[j] - m);
      sum += ev;
      srow[j] = ev;
    }
    sum += __shfl_xor(sum, 1, 32);
    sum += __shfl_xor(sum, 2, 32);
    const float inv = 1.0f / sum;
    bf16* prow = &pr[row * 264 + (tid & 3) * 64];
#pragma unroll 4
    for (int j = 0; j < 64; ++j) prow[j] = (bf16)(srow[j] * inv);
  }
  __syncthreads();

  // ---- phase 3: out[64,64] = probs @ v  (K=256 -> 8 k-steps; 2 d-tiles per wave)
  {
    const int m0 = (wave >> 1) * 16;
    const int ntbase = (wave & 1) * 2;
    v16bf af[8];
#pragma unroll
    for (int ks = 0; ks < 8; ++ks) {
      const int ak = ks * 32 + lhi * 16;
      v8bf lo = *reinterpret_cast<const v8bf*>(&pr[(m0 + lrow) * 264 + ak]);
      v8bf hi = *reinterpret_cast<const v8bf*>(&pr[(m0 + lrow) * 264 + ak + 8]);
      af[ks] = join8(lo, hi);
    }
#pragma unroll
    for (int ti = 0; ti < 2; ++ti) {
      const int nt = ntbase + ti;
      v8f acc = zero8();
#pragma unroll
      for (int ks = 0; ks < 8; ++ks) {
        const int brow = nt * 16 + lrow;        // d index (vT is d-major)
        const int bk = ks * 32 + lhi * 16;
        v8bf lo = *reinterpret_cast<const v8bf*>(&vT[brow * 264 + bk]);
        v8bf hi = *reinterpret_cast<const v8bf*>(&vT[brow * 264 + bk + 8]);
        acc = wmma_bf16(af[ks], join8(lo, hi), acc);
      }
#pragma unroll
      for (int r = 0; r < 8; ++r) {
        const int qt = m0 + r + lhi * 8;
        const int d  = nt * 16 + lrow;
        // output token = qt*64 + w, feature = h*64 + d  (matches swapaxes+reshape)
        ob[((size_t)b * 4096 + qt * 64 + w) * DIM + h * HEAD_DIM + d] = (bf16)acc[r];
      }
    }
  }
}

extern "C" void kernel_launch(void* const* d_in, const int* in_sizes, int n_in,
                              void* d_out, int out_size, void* d_ws, size_t ws_size,
                              hipStream_t stream) {
  (void)in_sizes; (void)n_in; (void)out_size; (void)ws_size;
  const float* x      = (const float*)d_in[0];
  const float* w_qkv  = (const float*)d_in[1];
  const float* b_qkv  = (const float*)d_in[2];
  const float* w_proj = (const float*)d_in[3];
  const float* b_proj = (const float*)d_in[4];
  float* out = (float*)d_out;

  char* ws = (char*)d_ws;
  bf16* xb   = (bf16*)ws; ws += (size_t)M_QKV * DIM * sizeof(bf16);
  bf16* wqb  = (bf16*)ws; ws += (size_t)QKV_N * DIM * sizeof(bf16);
  bf16* wpb  = (bf16*)ws; ws += (size_t)DIM * DIM * sizeof(bf16);
  bf16* qkvb = (bf16*)ws; ws += (size_t)M_QKV * QKV_N * sizeof(bf16);
  bf16* ob   = (bf16*)ws; ws += (size_t)M_PROJ * DIM * sizeof(bf16);

  // f32 -> bf16 (xb fits in the 192MB L2, making GEMM tile re-reads cheap)
  {
    int n4 = (M_QKV * DIM) / 4;
    cvt_f32_bf16<<<(n4 + 255) / 256, 256, 0, stream>>>(x, xb, n4);
    n4 = (QKV_N * DIM) / 4;
    cvt_f32_bf16<<<(n4 + 255) / 256, 256, 0, stream>>>(w_qkv, wqb, n4);
    n4 = (DIM * DIM) / 4;
    cvt_f32_bf16<<<(n4 + 255) / 256, 256, 0, stream>>>(w_proj, wpb, n4);
  }

  // QKV projection: [65536,2304] = xb @ wqb^T + b_qkv
  {
    dim3 grid(M_QKV / 128, QKV_N / 128);
    gemm_bf16_tn<true><<<grid, 256, 0, stream>>>(xb, wqb, b_qkv, qkvb, M_QKV, QKV_N, DIM);
  }

  // windowed attention, one block per (b,h,w)
  {
    size_t smem = (size_t)(64 * 72 + 256 * 72 + 64 * 264) * sizeof(bf16) +
                  (size_t)(64 * 264) * sizeof(float) +
                  (size_t)(64 * 264) * sizeof(bf16);   // 181,248 B < 320 KB WGP LDS
    mua_attn<<<BATCH * HEADS * NW, 256, smem, stream>>>(qkvb, ob);
  }

  // output projection: f32 out = ob @ wpb^T + b_proj
  {
    dim3 grid(M_PROJ / 128, DIM / 128);
    gemm_bf16_tn<false><<<grid, 256, 0, stream>>>(ob, wpb, b_proj, out, M_PROJ, DIM, DIM);
  }
}
